// MLA_21500606283818
// MI455X (gfx1250) — matrix operand order repaired and checked by reading, not compiled, not run on hardware
//
#include <hip/hip_runtime.h>
#include <hip/hip_bf16.h>
#include <stdint.h>

typedef __bf16 bf16;
typedef __attribute__((ext_vector_type(16))) __bf16 v16bf;
typedef __attribute__((ext_vector_type(8)))  __bf16 v8bf;
typedef __attribute__((ext_vector_type(8)))  float  v8f;

#define DEV static __device__ __forceinline__

static constexpr int BB   = 2;
static constexpr int SS   = 2048;
static constexpr int DIMX = 2048;
static constexpr int NHX  = 16;
static constexpr int DN   = 128;   // d_nope
static constexpr int DR   = 64;    // d_rope
static constexpr int DVx  = 128;   // d_v
static constexpr int DQK  = 192;
static constexpr int RANK = 512;

// ---------------- CDNA5 async global->LDS (ASYNCcnt) ------------------------
// GV mode: per-lane 16B DMA from the lane's global address to its LDS address.
// Generic pointers to __shared__ carry the LDS byte offset in the low 32 bits.

DEV void async_load_b128(void* lds_ptr, const void* gptr) {
  uint32_t lds_addr = (uint32_t)(uintptr_t)lds_ptr;
  asm volatile("global_load_async_to_lds_b128 %0, %1, off"
               :: "v"(lds_addr), "v"(gptr) : "memory");
}
DEV void wait_asynccnt0() {
  asm volatile("s_wait_asynccnt 0" ::: "memory");
}

// ---------------- WMMA fragment helpers (CDNA5 16x16x32 bf16 layouts) -------

// A-matrix 16x32 (MxK), row-major source with 'stride' halfwords between rows.
// lane 0-15: row=lane,    K = 0..7 then 16..23
// lane16-31: row=lane-16, K = 8..15 then 24..31
DEV v16bf load_a_frag(const bf16* base, int stride) {
  const int l   = threadIdx.x & 31;
  const int row = l & 15;
  const int kb  = (l >> 4) * 8;
  const bf16* p = base + row * stride + kb;
  v8bf lo = *(const v8bf*)(p);
  v8bf hi = *(const v8bf*)(p + 16);
  v16bf f;
#pragma unroll
  for (int i = 0; i < 8; ++i) { f[i] = lo[i]; f[i + 8] = hi[i]; }
  return f;
}

// B-matrix 32x16 (KxN) where B(k,n) = Wrow[n][k]; W row-major, 'stride' halfwords.
// lane 0-15: N=lane, K=0..7 then 16..23 ; lane 16-31: N=lane-16, K=8..15 then 24..31
DEV v16bf load_b_frag(const bf16* base, int stride) {
  const int l  = threadIdx.x & 31;
  const int n  = l & 15;
  const int kb = (l >> 4) * 8;
  const bf16* p = base + n * stride + kb;
  v8bf lo = *(const v8bf*)(p);
  v8bf hi = *(const v8bf*)(p + 16);
  v16bf f;
#pragma unroll
  for (int i = 0; i < 8; ++i) { f[i] = lo[i]; f[i + 8] = hi[i]; }
  return f;
}

DEV v8f wmma_bf16(v16bf a, v16bf b, v8f c) {
  return __builtin_amdgcn_wmma_f32_16x16x32_bf16(
      /*neg_a=*/false, a, /*neg_b=*/false, b,
      /*c_mod=*/(short)0, c, /*reuse_a=*/false, /*reuse_b=*/false);
}

// ---------------- fp32 -> bf16 conversion -----------------------------------

__global__ void to_bf16_kernel(const float* __restrict__ in, bf16* __restrict__ out,
                               size_t n) {
  size_t i = (size_t)blockIdx.x * blockDim.x + threadIdx.x;
  if (i < n) out[i] = (bf16)in[i];
}

// ---------------- generic GEMM: C[M,N] = A[M,K] * W[N,K]^T ------------------
// Block tile 128x128, BK=32; 256 threads = 8 waves; wave tile 32x64 =
// 2 A-frags x 4 B-frags -> 8 WMMAs per K-step.  LDS double-buffered with
// ASYNC global->LDS DMA: issue asyncs into `nxt`, run WMMAs from `cur`,
// s_wait_asynccnt 0, barrier.  DMA overlaps the matrix pipe.

template <typename OutT>
__global__ __launch_bounds__(256) void gemm_nt(const bf16* __restrict__ A,
                                               const bf16* __restrict__ W,
                                               OutT* __restrict__ C,
                                               int M, int N, int K) {
  __shared__ alignas(32) bf16 sA[2][128 * 32];
  __shared__ alignas(32) bf16 sW[2][128 * 32];
  const int tid  = threadIdx.x;
  const int wave = tid >> 5;
  const int m0 = blockIdx.y * 128;
  const int n0 = blockIdx.x * 128;
  const int wm = (wave & 3) * 32;    // 0,32,64,96
  const int wn = (wave >> 2) * 64;   // 0,64

  // staging: each thread DMAs 2x16B per tile (128 rows x 32 halves)
  const int r0 = tid >> 2;           // 0..63 (+64 for second half)
  const int c0 = (tid & 3) * 8;

  auto stage_async = [&](int buf, int kk) {
#pragma unroll
    for (int i = 0; i < 2; ++i) {
      int r = r0 + i * 64;
      async_load_b128(&sA[buf][r * 32 + c0], &A[(size_t)(m0 + r) * K + kk + c0]);
      int rw = n0 + r; rw = rw < N ? rw : N - 1;
      async_load_b128(&sW[buf][r * 32 + c0], &W[(size_t)rw * K + kk + c0]);
    }
  };

  v8f acc[2][4] = {};

  // prologue: DMA K-chunk 0 into buffer 0
  stage_async(0, 0);
  wait_asynccnt0();
  __syncthreads();

  const int nk = K / 32;
  for (int kt = 0; kt < nk; ++kt) {
    const int cur = kt & 1, nxt = cur ^ 1;
    const int k0 = kt * 32;
    const bool more = (kt + 1 < nk);
    if (more) {
      stage_async(nxt, k0 + 32);
      if (k0 + 64 < K)   // speculative prefetch of the K-slab after next
        __builtin_prefetch(&A[(size_t)(m0 + (tid >> 1)) * K + k0 + 64], 0, 1);
    }

    v16bf af[2], bfr[4];
#pragma unroll
    for (int i = 0; i < 2; ++i)
      af[i] = load_a_frag(&sA[cur][(wm + i * 16) * 32], 32);
#pragma unroll
    for (int j = 0; j < 4; ++j)
      bfr[j] = load_b_frag(&sW[cur][(wn + j * 16) * 32], 32);
#pragma unroll
    for (int i = 0; i < 2; ++i)
#pragma unroll
      for (int j = 0; j < 4; ++j)
        acc[i][j] = wmma_bf16(af[i], bfr[j], acc[i][j]);

    if (more) wait_asynccnt0();
    __syncthreads();
  }

  const int l     = tid & 31;
  const int rbase = (l >> 4) << 3;   // C layout: VGPR j -> row j (+8 hi half)
  const int col   = l & 15;
#pragma unroll
  for (int i = 0; i < 2; ++i)
#pragma unroll
    for (int j = 0; j < 4; ++j) {
      int cc = n0 + wn + j * 16 + col;
      if (cc < N) {
#pragma unroll
        for (int e = 0; e < 8; ++e)
          C[(size_t)(m0 + wm + i * 16 + rbase + e) * N + cc] = (OutT)acc[i][j][e];
      }
    }
}

// ---------------- rope on q_pe (last 64 dims of each 192 head-vec) ----------

__global__ void rope_q_kernel(bf16* __restrict__ q, const float* __restrict__ fc) {
  int idx = blockIdx.x * blockDim.x + threadIdx.x;        // (b,s,h,j) j<32
  if (idx >= BB * SS * NHX * (DR / 2)) return;
  int j = idx & 31;
  int s = (idx >> 9) & (SS - 1);
  size_t base = (size_t)(idx >> 5) * DQK + DN + 2 * j;
  float c  = fc[s * DR + 2 * j];
  float sn = fc[s * DR + 2 * j + 1];
  float x0 = (float)q[base], x1 = (float)q[base + 1];
  q[base]     = (bf16)(x0 * c - x1 * sn);
  q[base + 1] = (bf16)(x0 * sn + x1 * c);
}

// ---------------- RMS-norm(kv_lat) + rope(k_pe), one block per (b,s) --------

__global__ __launch_bounds__(256) void kv_proc_kernel(const float* __restrict__ kvf,
                                                      const float* __restrict__ w,
                                                      const float* __restrict__ fc,
                                                      bf16* __restrict__ kvn,
                                                      bf16* __restrict__ kpe) {
  __shared__ float red[8];
  const int row = blockIdx.x;                 // b*S + s
  const float* kv = kvf + (size_t)row * (RANK + DR);
  const int tid = threadIdx.x;
  float ss = 0.f;
  for (int i = tid; i < RANK; i += 256) { float v = kv[i]; ss += v * v; }
#pragma unroll
  for (int d = 1; d < 32; d <<= 1) ss += __shfl_xor(ss, d, 32);
  if ((tid & 31) == 0) red[tid >> 5] = ss;
  __syncthreads();
  if (tid < 8) {
    float v = red[tid];
#pragma unroll
    for (int d = 1; d < 8; d <<= 1) v += __shfl_xor(v, d, 32);
    if (tid == 0) red[0] = v;
  }
  __syncthreads();
  const float inv = rsqrtf(red[0] / RANK + 1e-6f);
  for (int i = tid; i < RANK; i += 256)
    kvn[(size_t)row * RANK + i] = (bf16)(kv[i] * inv * w[i]);
  if (tid < DR / 2) {
    int s = row & (SS - 1);
    int j = tid;
    float c  = fc[s * DR + 2 * j];
    float sn = fc[s * DR + 2 * j + 1];
    float x0 = kv[RANK + 2 * j], x1 = kv[RANK + 2 * j + 1];
    kpe[(size_t)row * DR + 2 * j]     = (bf16)(x0 * c - x1 * sn);
    kpe[(size_t)row * DR + 2 * j + 1] = (bf16)(x0 * sn + x1 * c);
  }
}

// ---------------- repack kv_up -> K(192 = nope||pe) and V(128) per head -----

__global__ void repack_kernel(const bf16* __restrict__ kvup,
                              const bf16* __restrict__ kpe,
                              bf16* __restrict__ kfull, bf16* __restrict__ v) {
  size_t idx = (size_t)blockIdx.x * blockDim.x + threadIdx.x;
  if (idx >= (size_t)BB * SS * NHX * 256) return;
  int d = (int)(idx & 255);
  size_t row = idx >> 8;                      // (b,s,h) flat
  if (d < 128) {
    kfull[row * DQK + d] = kvup[idx];
    if (d < DR) kfull[row * DQK + DN + d] = kpe[(row >> 4) * DR + d];
  } else {
    v[row * DVx + (d - 128)] = kvup[idx];
  }
}

// ---------------- flash attention: 128 q-rows/block, 8 waves x 16 rows ------

__global__ __launch_bounds__(256) void mla_attn_kernel(const bf16* __restrict__ Q,
                                                       const bf16* __restrict__ Kf,
                                                       const bf16* __restrict__ V,
                                                       bf16* __restrict__ O,
                                                       float scale) {
  __shared__ alignas(32) bf16 sK[64 * DQK];     // 24 KB, row-major (t, 192)
  __shared__ alignas(32) bf16 sVt[DVx * 64];    // 16 KB, transposed (d, t)
  __shared__ alignas(32) bf16 sP[8][16 * 64];   // 16 KB, per-wave P tile
  const int tid  = threadIdx.x;
  const int wave = tid >> 5;
  const int l    = tid & 31;
  const int b = blockIdx.z, h = blockIdx.y;
  const int q0 = blockIdx.x * 128 + wave * 16;
  const int qstride = NHX * DQK;
  const bf16* qbase = Q + ((size_t)(b * SS + q0) * NHX + h) * DQK;

  v16bf qf[6];
#pragma unroll
  for (int c = 0; c < 6; ++c) qf[c] = load_a_frag(qbase + c * 32, qstride);

  v8f oacc[8] = {};
  float m[8], lsum[8];
#pragma unroll
  for (int j = 0; j < 8; ++j) { m[j] = -3.0e38f; lsum[j] = 0.f; }

  const int rbase = (l >> 4) << 3;
  const int col   = l & 15;
  const int nkb   = (blockIdx.x + 1) * 2;       // causal: keys up to (bx+1)*128

  for (int kb = 0; kb < nkb; ++kb) {
    const int t0 = kb * 64;
    __syncthreads();
    // stage K block (64 x 192 halfwords) via async global->LDS DMA
    for (int idx = tid; idx < 64 * 24; idx += 256) {
      int r = idx / 24, c8 = (idx % 24) * 8;
      async_load_b128(&sK[r * DQK + c8],
                      &Kf[((size_t)(b * SS + t0 + r) * NHX + h) * DQK + c8]);
    }
    // stage V transposed: coalesced uint4 global loads, LDS scatter
    // (overlaps with the async K DMA above)
    for (int idx = tid; idx < 64 * 16; idx += 256) {
      int t = idx >> 4, d8 = (idx & 15) * 8;
      uint4 pk = *(const uint4*)&V[((size_t)(b * SS + t0 + t) * NHX + h) * DVx + d8];
      bf16 tmp[8];
      *(uint4*)tmp = pk;
#pragma unroll
      for (int e = 0; e < 8; ++e) sVt[(d8 + e) * 64 + t] = tmp[e];
    }
    wait_asynccnt0();
    __syncthreads();

    // S = Q * K^T : 4 tiles of 16 keys, contraction over 192 in 6 chunks.
    v8f sc[4] = {};
#pragma unroll
    for (int c = 0; c < 6; ++c) {
      v16bf bt[4];
#pragma unroll
      for (int n = 0; n < 4; ++n)
        bt[n] = load_b_frag(sK + n * 16 * DQK + c * 32, DQK);
#pragma unroll
      for (int n = 0; n < 4; ++n)
        sc[n] = wmma_bf16(qf[c], bt[n], sc[n]);
    }

    // scale + causal mask + online softmax (row r lives in one 16-lane half)
    float p[4][8];
#pragma unroll
    for (int j = 0; j < 8; ++j) {
      float mj = -3.0e38f;
#pragma unroll
      for (int n = 0; n < 4; ++n) {
        float s = sc[n][j] * scale;
        int tabs = t0 + n * 16 + col;
        int rabs = q0 + rbase + j;
        s = (tabs > rabs) ? -3.0e38f : s;
        p[n][j] = s;
        mj = fmaxf(mj, s);
      }
#pragma unroll
      for (int d = 1; d < 16; d <<= 1) mj = fmaxf(mj, __shfl_xor(mj, d, 32));
      float mnew  = fmaxf(m[j], mj);
      float alpha = __expf(m[j] - mnew);
      float rsum  = 0.f;
#pragma unroll
      for (int n = 0; n < 4; ++n) {
        float e = __expf(p[n][j] - mnew);
        p[n][j] = e;
        rsum += e;
      }
#pragma unroll
      for (int d = 1; d < 16; d <<= 1) rsum += __shfl_xor(rsum, d, 32);
      lsum[j] = lsum[j] * alpha + rsum;
      m[j] = mnew;
#pragma unroll
      for (int nv = 0; nv < 8; ++nv) oacc[nv][j] *= alpha;
    }

    // round-trip P through LDS to convert C-layout -> A-fragment layout
#pragma unroll
    for (int n = 0; n < 4; ++n)
#pragma unroll
      for (int j = 0; j < 8; ++j)
        sP[wave][(rbase + j) * 64 + n * 16 + col] = (bf16)p[n][j];

    // O += P * V  (contraction over 64 keys in 2 chunks of 32)
#pragma unroll
    for (int c2 = 0; c2 < 2; ++c2) {
      v16bf pa = load_a_frag(&sP[wave][c2 * 32], 64);
#pragma unroll
      for (int g = 0; g < 2; ++g) {
        v16bf bv[4];
#pragma unroll
        for (int n = 0; n < 4; ++n)
          bv[n] = load_b_frag(sVt + (g * 4 + n) * 16 * 64 + c2 * 32, 64);
#pragma unroll
        for (int n = 0; n < 4; ++n)
          oacc[g * 4 + n] = wmma_bf16(pa, bv[n], oacc[g * 4 + n]);
      }
    }
  }

  // normalize and write O (b,s,h,128) bf16
#pragma unroll
  for (int j = 0; j < 8; ++j) {
    float inv = 1.0f / lsum[j];
    size_t r = (size_t)(b * SS + q0 + rbase + j) * NHX + h;
#pragma unroll
    for (int nv = 0; nv < 8; ++nv)
      O[r * DVx + nv * 16 + col] = (bf16)(oacc[nv][j] * inv);
  }
}

// ---------------- host-side orchestration -----------------------------------

extern "C" void kernel_launch(void* const* d_in, const int* in_sizes, int n_in,
                              void* d_out, int out_size, void* d_ws, size_t ws_size,
                              hipStream_t stream) {
  (void)in_sizes; (void)n_in; (void)out_size; (void)ws_size;
  const float* x   = (const float*)d_in[0];
  // d_in[1] = start_pos (0), d_in[3] = mask (causal, rebuilt in-kernel)
  const float* fc  = (const float*)d_in[2];
  const float* wq  = (const float*)d_in[4];
  const float* wa  = (const float*)d_in[5];
  const float* wb  = (const float*)d_in[6];
  const float* wo  = (const float*)d_in[7];
  const float* knw = (const float*)d_in[8];
  float* out = (float*)d_out;

  const int M = BB * SS;                       // 4096 token rows
  char* ws = (char*)d_ws;
  size_t off = 0;
  auto alloc = [&](size_t bytes) -> void* {
    void* p = ws + off;
    off = (off + bytes + 255) & ~(size_t)255;
    return p;
  };

  bf16* xb    = (bf16*)alloc((size_t)M * DIMX * 2);
  bf16* wqb   = (bf16*)alloc((size_t)NHX * DQK * DIMX * 2);
  bf16* wab   = (bf16*)alloc((size_t)(RANK + DR) * DIMX * 2);
  bf16* wbb   = (bf16*)alloc((size_t)NHX * (DN + DVx) * RANK * 2);
  bf16* wob   = (bf16*)alloc((size_t)DIMX * (NHX * DVx) * 2);
  bf16* qb    = (bf16*)alloc((size_t)M * NHX * DQK * 2);
  float* kvf  = (float*)alloc((size_t)M * (RANK + DR) * 4);
  bf16* kvn   = (bf16*)alloc((size_t)M * RANK * 2);
  bf16* kpe   = (bf16*)alloc((size_t)M * DR * 2);
  bf16* kvup  = (bf16*)alloc((size_t)M * NHX * 256 * 2);
  bf16* kfull = (bf16*)alloc((size_t)M * NHX * DQK * 2);
  bf16* vbuf  = (bf16*)alloc((size_t)M * NHX * DVx * 2);
  bf16* aout  = (bf16*)alloc((size_t)M * NHX * DVx * 2);

  auto cvt = [&](const float* src, bf16* dst, size_t n) {
    to_bf16_kernel<<<(unsigned)((n + 255) / 256), 256, 0, stream>>>(src, dst, n);
  };
  cvt(x,  xb,  (size_t)M * DIMX);
  cvt(wq, wqb, (size_t)NHX * DQK * DIMX);
  cvt(wa, wab, (size_t)(RANK + DR) * DIMX);
  cvt(wb, wbb, (size_t)NHX * (DN + DVx) * RANK);
  cvt(wo, wob, (size_t)DIMX * (NHX * DVx));

  // q = x @ wq^T  : (4096 x 3072), K=2048
  gemm_nt<bf16><<<dim3((NHX * DQK + 127) / 128, M / 128), 256, 0, stream>>>(
      xb, wqb, qb, M, NHX * DQK, DIMX);
  rope_q_kernel<<<(M * NHX * (DR / 2) + 255) / 256, 256, 0, stream>>>(qb, fc);

  // kv = x @ wkv_a^T : (4096 x 576) f32, K=2048
  gemm_nt<float><<<dim3((RANK + DR + 127) / 128, M / 128), 256, 0, stream>>>(
      xb, wab, kvf, M, RANK + DR, DIMX);
  kv_proc_kernel<<<M, 256, 0, stream>>>(kvf, knw, fc, kvn, kpe);

  // kv_up = rmsnorm(kv_lat) @ wkv_b^T : (4096 x 4096), K=512
  gemm_nt<bf16><<<dim3((NHX * 256 + 127) / 128, M / 128), 256, 0, stream>>>(
      kvn, wbb, kvup, M, NHX * 256, RANK);
  repack_kernel<<<(int)(((size_t)M * NHX * 256 + 255) / 256), 256, 0, stream>>>(
      kvup, kpe, kfull, vbuf);

  // attention
  const float scale = 0.07216878364870322f;     // 192^-0.5
  mla_attn_kernel<<<dim3(SS / 128, NHX, BB), 256, 0, stream>>>(
      qb, kfull, vbuf, aout, scale);

  // out = attn @ wo^T : (4096 x 2048), K=2048, f32 output
  gemm_nt<float><<<dim3((DIMX + 127) / 128, M / 128), 256, 0, stream>>>(
      aout, wob, out, M, DIMX, NHX * DVx);
}
